// LargeLanguageModel_36112085024773
// MI455X (gfx1250) — compile-verified, matrix operand order checked by reading, not run on hardware
//
#include <hip/hip_runtime.h>

// ---------------------------------------------------------------------------
// GPT forward for MI455X (gfx1250, wave32, WMMA).
// Compute-bound (~445 GFLOP vs ~500MB weights) => all GEMMs run
// v_wmma_f32_16x16x32_f16 with f32 accumulation. Operand tiles are staged
// k-contiguous so every fragment is exactly 2x ds_load_b128 (weights
// pre-transposed to [N][K] f16 once per call; V transposed by the projection
// epilogue). Main GEMMs use double-buffered LDS so global loads + ds stores
// overlap the WMMA stream, plus global_prefetch_b8 at distance-2 tiles.
// ---------------------------------------------------------------------------

typedef __attribute__((ext_vector_type(16))) _Float16 v16h;
typedef __attribute__((ext_vector_type(8)))  _Float16 v8h;
typedef __attribute__((ext_vector_type(8)))  float    v8f;

#define DEVINL __device__ __forceinline__

constexpr int kV  = 32000;
constexpr int kC  = 1024;
constexpr int kH  = 16;
constexpr int kL  = 4;
constexpr int kB  = 4;
constexpr int kT  = 512;
constexpr int kFF = 4 * kC;       // 4096
constexpr int kHD = kC / kH;      // 64
constexpr int kM  = kB * kT;      // 2048 token rows

// Macro-tiles. Generic GEMM / AV: 128x64 per 128-thread block (4 waves, each
// wave: 32 rows x 64 cols = 8 WMMA per K-step). Scores: 64x64.
constexpr int GM = 128;
constexpr int BM = 64, BN = 64, BK = 32;
constexpr int SA = BK + 8;        // LDS row stride in halves (80B: 16B-aligned)

DEVINL float neg_inf() { return -__builtin_inff(); }

// ---------------------------------------------------------------------------
// Block reductions (wave32)
// ---------------------------------------------------------------------------
template <int NW>
DEVINL float block_sum(float v, float* red, int tid) {
#pragma unroll
  for (int o = 16; o > 0; o >>= 1) v += __shfl_xor(v, o, 32);
  __syncthreads();
  if ((tid & 31) == 0) red[tid >> 5] = v;
  __syncthreads();
  float t = 0.f;
#pragma unroll
  for (int i = 0; i < NW; ++i) t += red[i];
  return t;
}

template <int NW>
DEVINL float block_max(float v, float* red, int tid) {
#pragma unroll
  for (int o = 16; o > 0; o >>= 1) v = fmaxf(v, __shfl_xor(v, o, 32));
  __syncthreads();
  if ((tid & 31) == 0) red[tid >> 5] = v;
  __syncthreads();
  float t = red[0];
#pragma unroll
  for (int i = 1; i < NW; ++i) t = fmaxf(t, red[i]);
  return t;
}

// ---------------------------------------------------------------------------
// Fragment load per CDNA5 ISA 7.12.2: for 16-bit 16x32 operands, lane (row,
// lhi) needs k = lhi*8..+7 then 16+lhi*8..+7 -> two contiguous 8-half runs ->
// two ds_load_b128 from a k-contiguous LDS tile row.
// ---------------------------------------------------------------------------
DEVINL v16h load_frag(const _Float16* tile, int row, int lhi) {
  const _Float16* p = tile + row * SA;
  const v8h lo = *(const v8h*)(p + (lhi << 3));
  const v8h hi = *(const v8h*)(p + 16 + (lhi << 3));
  return __builtin_shufflevector(lo, hi, 0, 1, 2, 3, 4, 5, 6, 7, 8, 9, 10, 11,
                                 12, 13, 14, 15);
}

DEVINL v8f wmma_f16(v16h a, v16h b, v8f c) {
  return __builtin_amdgcn_wmma_f32_16x16x32_f16(false, a, false, b, (short)0, c,
                                                false, false);
}

// ---------------------------------------------------------------------------
// Weight convert+transpose: out[n*K + k] = (f16)in[k*N + n]. Coalesced both
// directions via 32x32 LDS tile. One-time ~180MB => ~8us at 23.3 TB/s.
// ---------------------------------------------------------------------------
__global__ void __launch_bounds__(256)
k_wcvt_t(const float* __restrict__ in, _Float16* __restrict__ out, int K, int N) {
  __shared__ _Float16 tile[32][33];
  const int lx = threadIdx.x & 31, ly = threadIdx.x >> 5;
  const int nBase = blockIdx.x * 32, kBase = blockIdx.y * 32;
#pragma unroll
  for (int r = ly; r < 32; r += 8)
    tile[r][lx] = (_Float16)in[(size_t)(kBase + r) * N + nBase + lx];
  __syncthreads();
#pragma unroll
  for (int r = ly; r < 32; r += 8)
    out[(size_t)(nBase + r) * K + kBase + lx] = tile[lx][r];
}

// ---------------------------------------------------------------------------
// Embedding: h[b,t,:] = tok[x[b,t],:] + pos[t,:]
// ---------------------------------------------------------------------------
__global__ void k_embed(const int* __restrict__ x, const float* __restrict__ tok,
                        const float* __restrict__ pos, float* __restrict__ h) {
  const int row = blockIdx.x;
  const int t   = row & (kT - 1);
  const int id  = x[row];
  const float* tp = tok + (size_t)id * kC;
  const float* pp = pos + (size_t)t * kC;
  float* hp = h + (size_t)row * kC;
  for (int c = threadIdx.x; c < kC; c += blockDim.x) hp[c] = tp[c] + pp[c];
}

// ---------------------------------------------------------------------------
// LayerNorm over C=1024; optional f32 out (may alias input) + optional f16 out
// ---------------------------------------------------------------------------
__global__ void __launch_bounds__(256)
k_layernorm(const float* __restrict__ in, const float* __restrict__ g,
            const float* __restrict__ b, float* __restrict__ out32,
            _Float16* __restrict__ out16) {
  constexpr int NT = 256;
  __shared__ float red[NT / 32];
  const int row = blockIdx.x, tid = threadIdx.x;
  const float* xr = in + (size_t)row * kC;

  float lx[kC / NT];
  float s = 0.f;
#pragma unroll
  for (int i = 0; i < kC / NT; ++i) { lx[i] = xr[tid + i * NT]; s += lx[i]; }
  const float mean = block_sum<NT / 32>(s, red, tid) * (1.f / kC);

  float vs = 0.f;
#pragma unroll
  for (int i = 0; i < kC / NT; ++i) { const float d = lx[i] - mean; vs += d * d; }
  const float var  = block_sum<NT / 32>(vs, red, tid) * (1.f / kC);
  const float rstd = rsqrtf(var + 1e-5f);

#pragma unroll
  for (int i = 0; i < kC / NT; ++i) {
    const int c = tid + i * NT;
    const float o = (lx[i] - mean) * rstd * g[c] + b[c];
    if (out32) out32[(size_t)row * kC + c] = o;
    if (out16) out16[(size_t)row * kC + c] = (_Float16)o;
  }
}

// ---------------------------------------------------------------------------
// Generic WMMA GEMM: out[M,N] = A[M,K](f16 row-major) x W, W stored TRANSPOSED
// as BT[N,K] (f16). 128x64 tile, 4 waves, 8 WMMA/wave/K-step, double-buffered
// LDS: compute tile i from buf i%2 while tile i+1 streams into buf (i+1)%2.
// Epilogue: (+bias) (ReLU) (+res) -> f32 / f16 / f16-transposed-V stores.
// ---------------------------------------------------------------------------
__global__ void __launch_bounds__(128)
k_gemm_f16(const _Float16* __restrict__ A, const _Float16* __restrict__ BT,
           const float* __restrict__ bias, const float* __restrict__ res,
           float* __restrict__ out32, _Float16* __restrict__ out16,
           _Float16* __restrict__ out16vt,   // V layout: [(b*H+h)*HD+d][T]
           int N, int K, int relu) {
  __shared__ alignas(16) _Float16 As[2][GM * SA];   // [m][k]
  __shared__ alignas(16) _Float16 Bs[2][BN * SA];   // [n][k]
  const int tid  = threadIdx.x;
  const int warp = tid >> 5, lane = tid & 31;
  const int lrow = lane & 15, lhi = lane >> 4;
  const int rowBase = blockIdx.y * GM, colBase = blockIdx.x * BN;

  // per-thread staging addresses (k0 added per iteration)
  const _Float16* agp[4];
  int aoff[4];
#pragma unroll
  for (int it = 0; it < 4; ++it) {
    const int idx = (tid + it * 128) * 8;
    const int r = idx >> 5, c = idx & 31;
    agp[it]  = A + (size_t)(rowBase + r) * K + c;
    aoff[it] = r * SA + c;
  }
  const _Float16* bgp[2];
  int boff[2];
#pragma unroll
  for (int it = 0; it < 2; ++it) {
    const int idx = (tid + it * 128) * 8;
    const int r = idx >> 5, c = idx & 31;
    bgp[it]  = BT + (size_t)(colBase + r) * K + c;
    boff[it] = r * SA + c;
  }

  // prologue: stage tile 0 into buffer 0
#pragma unroll
  for (int it = 0; it < 4; ++it) *(v8h*)(&As[0][aoff[it]]) = *(const v8h*)(agp[it]);
#pragma unroll
  for (int it = 0; it < 2; ++it) *(v8h*)(&Bs[0][boff[it]]) = *(const v8h*)(bgp[it]);
  __syncthreads();

  v8f acc[2][4] = {};
  int buf = 0;
  for (int k0 = 0; k0 < K; k0 += BK, buf ^= 1) {
    const bool more = (k0 + BK) < K;
    v8h ra[4], rb[2];
    if (more) {
#pragma unroll
      for (int it = 0; it < 4; ++it) ra[it] = *(const v8h*)(agp[it] + k0 + BK);
#pragma unroll
      for (int it = 0; it < 2; ++it) rb[it] = *(const v8h*)(bgp[it] + k0 + BK);
      if (k0 + 2 * BK < K) {              // warm L2/WGP$ two tiles ahead
        __builtin_prefetch(agp[0] + k0 + 2 * BK, 0, 0);
        __builtin_prefetch(bgp[0] + k0 + 2 * BK, 0, 0);
      }
    }
    const _Float16* as = As[buf];
    const _Float16* bs = Bs[buf];
    const v16h a0 = load_frag(as, warp * 32 + lrow, lhi);
    const v16h a1 = load_frag(as, warp * 32 + 16 + lrow, lhi);
#pragma unroll
    for (int j = 0; j < 4; ++j) {
      const v16h bf = load_frag(bs, j * 16 + lrow, lhi);
      acc[0][j] = wmma_f16(a0, bf, acc[0][j]);
      acc[1][j] = wmma_f16(a1, bf, acc[1][j]);
    }
    if (more) {
#pragma unroll
      for (int it = 0; it < 4; ++it) *(v8h*)(&As[buf ^ 1][aoff[it]]) = ra[it];
#pragma unroll
      for (int it = 0; it < 2; ++it) *(v8h*)(&Bs[buf ^ 1][boff[it]]) = rb[it];
      __syncthreads();
    }
  }

#pragma unroll
  for (int fi = 0; fi < 2; ++fi) {
#pragma unroll
    for (int j = 0; j < 4; ++j) {
      const int col = colBase + j * 16 + lrow;
      const float bv = bias ? bias[col] : 0.f;
#pragma unroll
      for (int i = 0; i < 8; ++i) {
        const int row = rowBase + warp * 32 + fi * 16 + lhi * 8 + i;
        const size_t o = (size_t)row * N + col;
        float v = acc[fi][j][i] + bv;
        if (relu) v = fmaxf(v, 0.f);
        if (res) v += res[o];
        if (out32) out32[o] = v;
        if (out16) out16[o] = (_Float16)v;
        if (out16vt)  // row = b*T+t, col = h*64+d  ->  [(b*C+col)][t]
          out16vt[((size_t)(row >> 9) * kC + col) * kT + (row & (kT - 1))] =
              (_Float16)v;
      }
    }
  }
}

// ---------------------------------------------------------------------------
// scores[bh,t,s] = (q . k) / sqrt(C), causal. q,k are f16 [M,C] (head slice at
// column h*64); key rows are already n-major. Only 2 K-steps (hd=64).
// ---------------------------------------------------------------------------
__global__ void __launch_bounds__(128)
k_attn_scores(const _Float16* __restrict__ q, const _Float16* __restrict__ kmat,
              float* __restrict__ scores) {
  const int bh = blockIdx.z, b = bh >> 4, hh = bh & 15;
  const int tBase = blockIdx.y * BM, sBase = blockIdx.x * BN;
  const int tid = threadIdx.x, warp = tid >> 5, lane = tid & 31;
  const int lrow = lane & 15, lhi = lane >> 4;
  float* srow = scores + (size_t)bh * kT * kT;

  if (sBase > tBase + (BM - 1)) {                 // tile fully above diagonal
    for (int idx = tid; idx < BM * BN; idx += 128) {
      const int r = idx >> 6, c2 = idx & 63;
      srow[(size_t)(tBase + r) * kT + sBase + c2] = neg_inf();
    }
    return;
  }

  __shared__ alignas(16) _Float16 As[BM * SA];
  __shared__ alignas(16) _Float16 Bs[BN * SA];
  const _Float16* qb = q    + ((size_t)b * kT) * kC + hh * kHD;
  const _Float16* kb = kmat + ((size_t)b * kT) * kC + hh * kHD;

  v8f acc[4] = {};
  for (int k0 = 0; k0 < kHD; k0 += BK) {          // 2 iterations
    __syncthreads();
#pragma unroll
    for (int it = 0; it < 2; ++it) {
      const int idx = (tid + it * 128) * 8;
      const int r = idx >> 5, c = idx & 31;
      *(v8h*)(As + r * SA + c) =
          *(const v8h*)(qb + (size_t)(tBase + r) * kC + k0 + c);
      *(v8h*)(Bs + r * SA + c) =
          *(const v8h*)(kb + (size_t)(sBase + r) * kC + k0 + c);
    }
    __syncthreads();
    const v16h a = load_frag(As, warp * 16 + lrow, lhi);
#pragma unroll
    for (int j = 0; j < 4; ++j) {
      const v16h bf = load_frag(Bs, j * 16 + lrow, lhi);
      acc[j] = wmma_f16(a, bf, acc[j]);
    }
  }

  const float scale = 1.0f / 32.0f;               // 1/sqrt(C), per reference
#pragma unroll
  for (int j = 0; j < 4; ++j) {
    const int s = sBase + j * 16 + lrow;
#pragma unroll
    for (int i = 0; i < 8; ++i) {
      const int t = tBase + warp * 16 + lhi * 8 + i;
      srow[(size_t)t * kT + s] = (s <= t) ? acc[j][i] * scale : neg_inf();
    }
  }
}

// ---------------------------------------------------------------------------
// Row softmax over T=512 (-inf -> 0), writes P as f16
// ---------------------------------------------------------------------------
__global__ void __launch_bounds__(128)
k_softmax_rows(const float* __restrict__ scores, _Float16* __restrict__ p) {
  constexpr int NT = 128;
  __shared__ float red[NT / 32];
  const size_t row = blockIdx.x;                  // B*H*T rows
  const float* sr = scores + row * kT;
  _Float16* pr = p + row * kT;
  const int tid = threadIdx.x;

  float lv[kT / NT];
  float mx = neg_inf();
#pragma unroll
  for (int i = 0; i < kT / NT; ++i) { lv[i] = sr[tid + i * NT]; mx = fmaxf(mx, lv[i]); }
  mx = block_max<NT / 32>(mx, red, tid);

  float s = 0.f;
#pragma unroll
  for (int i = 0; i < kT / NT; ++i) { lv[i] = expf(lv[i] - mx); s += lv[i]; }
  s = block_sum<NT / 32>(s, red, tid);
  const float inv = 1.0f / s;
#pragma unroll
  for (int i = 0; i < kT / NT; ++i) pr[tid + i * NT] = (_Float16)(lv[i] * inv);
}

// ---------------------------------------------------------------------------
// O[b,t,h*64+d] = sum_s P[bh,t,s] * V[b,s,h,d] (+res). V pre-transposed to
// vT[(b*H+h)*64+d][T] so the B tile is [n][k]. 128x64 tile, double-buffered;
// K-loop clipped by causality (P[t,s]=0 for s>t).
// ---------------------------------------------------------------------------
__global__ void __launch_bounds__(128)
k_attn_av(const _Float16* __restrict__ p, const _Float16* __restrict__ vT,
          const float* __restrict__ res, float* __restrict__ out) {
  const int bh = blockIdx.y, b = bh >> 4, hh = bh & 15;
  const int tBase = blockIdx.x * GM;
  const int tid = threadIdx.x, warp = tid >> 5, lane = tid & 31;
  const int lrow = lane & 15, lhi = lane >> 4;

  __shared__ alignas(16) _Float16 As[2][GM * SA];   // P tile [t][s]
  __shared__ alignas(16) _Float16 Bs[2][BN * SA];   // V tile [d][s]
  const _Float16* pb = p + (size_t)bh * kT * kT;
  const _Float16* vb = vT + (size_t)bh * kHD * kT;

  const _Float16* agp[4];
  int aoff[4];
#pragma unroll
  for (int it = 0; it < 4; ++it) {
    const int idx = (tid + it * 128) * 8;
    const int r = idx >> 5, c = idx & 31;
    agp[it]  = pb + (size_t)(tBase + r) * kT + c;
    aoff[it] = r * SA + c;
  }
  const _Float16* bgp[2];
  int boff[2];
#pragma unroll
  for (int it = 0; it < 2; ++it) {
    const int idx = (tid + it * 128) * 8;
    const int r = idx >> 5, c = idx & 31;
    bgp[it]  = vb + (size_t)r * kT + c;
    boff[it] = r * SA + c;
  }

#pragma unroll
  for (int it = 0; it < 4; ++it) *(v8h*)(&As[0][aoff[it]]) = *(const v8h*)(agp[it]);
#pragma unroll
  for (int it = 0; it < 2; ++it) *(v8h*)(&Bs[0][boff[it]]) = *(const v8h*)(bgp[it]);
  __syncthreads();

  v8f acc[2][4] = {};
  const int kmax = tBase + GM;                    // causal extent
  int buf = 0;
  for (int k0 = 0; k0 < kmax; k0 += BK, buf ^= 1) {
    const bool more = (k0 + BK) < kmax;
    v8h ra[4], rb[2];
    if (more) {
#pragma unroll
      for (int it = 0; it < 4; ++it) ra[it] = *(const v8h*)(agp[it] + k0 + BK);
#pragma unroll
      for (int it = 0; it < 2; ++it) rb[it] = *(const v8h*)(bgp[it] + k0 + BK);
    }
    const _Float16* as = As[buf];
    const _Float16* bs = Bs[buf];
    const v16h a0 = load_frag(as, warp * 32 + lrow, lhi);
    const v16h a1 = load_frag(as, warp * 32 + 16 + lrow, lhi);
#pragma unroll
    for (int j = 0; j < 4; ++j) {
      const v16h bf = load_frag(bs, j * 16 + lrow, lhi);
      acc[0][j] = wmma_f16(a0, bf, acc[0][j]);
      acc[1][j] = wmma_f16(a1, bf, acc[1][j]);
    }
    if (more) {
#pragma unroll
      for (int it = 0; it < 4; ++it) *(v8h*)(&As[buf ^ 1][aoff[it]]) = ra[it];
#pragma unroll
      for (int it = 0; it < 2; ++it) *(v8h*)(&Bs[buf ^ 1][boff[it]]) = rb[it];
      __syncthreads();
    }
  }

#pragma unroll
  for (int fi = 0; fi < 2; ++fi) {
#pragma unroll
    for (int j = 0; j < 4; ++j) {
      const int d = hh * kHD + j * 16 + lrow;
#pragma unroll
      for (int i = 0; i < 8; ++i) {
        const int t = tBase + warp * 32 + fi * 16 + lhi * 8 + i;
        const size_t o = ((size_t)b * kT + t) * kC + d;
        out[o] = acc[fi][j][i] + res[o];
      }
    }
  }
}

// ---------------------------------------------------------------------------
// Host driver
// ---------------------------------------------------------------------------
extern "C" void kernel_launch(void* const* d_in, const int* in_sizes, int n_in,
                              void* d_out, int out_size, void* d_ws, size_t ws_size,
                              hipStream_t stream) {
  (void)in_sizes; (void)n_in; (void)out_size; (void)ws_size;

  const int*   x   = (const int*)d_in[0];
  const float* tok = (const float*)d_in[1];
  const float* pos = (const float*)d_in[2];
  auto LYR = [&](int l, int j) -> const float* {
    return (const float*)d_in[3 + 20 * l + j];
  };
  const float* lnf_g = (const float*)d_in[3 + 20 * kL + 0];
  const float* lnf_b = (const float*)d_in[3 + 20 * kL + 1];
  const float* Whead = (const float*)d_in[3 + 20 * kL + 2];
  const float* bhead = (const float*)d_in[3 + 20 * kL + 3];

  // ---- workspace carve (256B aligned) ----
  char* wp = (char*)d_ws;
  auto a16 = [&](size_t n) {
    _Float16* p = (_Float16*)wp; wp += ((n * 2 + 255) & ~(size_t)255); return p;
  };
  auto a32 = [&](size_t n) {
    float* p = (float*)wp; wp += ((n * 4 + 255) & ~(size_t)255); return p;
  };

  _Float16* w16[kL][8];   // transposed f16: Wq1,Wk1,Wv1,Wq2,Wk2,Wv2,W1,W2
  for (int l = 0; l < kL; ++l) {
    for (int j = 0; j < 6; ++j) w16[l][j] = a16((size_t)kC * kC);
    w16[l][6] = a16((size_t)kC * kFF);
    w16[l][7] = a16((size_t)kFF * kC);
  }
  _Float16* whead16 = a16((size_t)kC * kV);
  float*    Abuf = a32((size_t)kM * kC);            // h
  float*    Bbuf = a32((size_t)kM * kC);            // y
  float*    Cbuf = a32((size_t)kM * kC);            // z
  _Float16* ln16 = a16((size_t)kM * kC);
  _Float16* q16  = a16((size_t)kM * kC);
  _Float16* k16  = a16((size_t)kM * kC);
  _Float16* vT16 = a16((size_t)kM * kC);            // [(b*H+h)*64+d][T]
  _Float16* ffh16 = a16((size_t)kM * kFF);
  float*    scores = a32((size_t)kB * kH * kT * kT);
  _Float16* p16    = a16((size_t)kB * kH * kT * kT);

  // weights: f32[K][N] -> f16[N][K]
  auto convT = [&](const float* src, _Float16* dst, int K, int N) {
    k_wcvt_t<<<dim3(N / 32, K / 32), 256, 0, stream>>>(src, dst, K, N);
  };
  static const int widx[8] = {2, 3, 4, 9, 10, 11, 14, 16};
  for (int l = 0; l < kL; ++l)
    for (int j = 0; j < 8; ++j) {
      const int K = (j == 7) ? kFF : kC;
      const int N = (j == 6) ? kFF : kC;
      convT(LYR(l, widx[j]), w16[l][j], K, N);
    }
  convT(Whead, whead16, kC, kV);

  auto ln = [&](const float* in, const float* g, const float* b, float* o32,
                _Float16* o16) {
    k_layernorm<<<kM, 256, 0, stream>>>(in, g, b, o32, o16);
  };
  auto gemm = [&](const _Float16* A, const _Float16* BT, const float* bias,
                  const float* res, float* o32, _Float16* o16, _Float16* o16vt,
                  int N, int K, int relu) {
    k_gemm_f16<<<dim3(N / BN, kM / GM), 128, 0, stream>>>(A, BT, bias, res, o32,
                                                          o16, o16vt, N, K, relu);
  };
  // attention(LN(hin, g, b)) + hin -> outbuf
  auto attention = [&](const float* hin, const float* g, const float* b,
                       _Float16* wq, _Float16* wk, _Float16* wv, float* outbuf) {
    ln(hin, g, b, nullptr, ln16);
    gemm(ln16, wq, nullptr, nullptr, nullptr, q16, nullptr, kC, kC, 0);
    gemm(ln16, wk, nullptr, nullptr, nullptr, k16, nullptr, kC, kC, 0);
    gemm(ln16, wv, nullptr, nullptr, nullptr, nullptr, vT16, kC, kC, 0);
    k_attn_scores<<<dim3(kT / BN, kT / BM, kB * kH), 128, 0, stream>>>(q16, k16,
                                                                       scores);
    k_softmax_rows<<<kB * kH * kT, 128, 0, stream>>>(scores, p16);
    k_attn_av<<<dim3(kT / GM, kB * kH), 128, 0, stream>>>(p16, vT16, hin, outbuf);
  };

  // h = tok[x] + pos
  k_embed<<<kM, 256, 0, stream>>>(x, tok, pos, Abuf);

  float* hbuf = Abuf;
  for (int l = 0; l < kL; ++l) {
    // y = attn1(LN(h, ln1)) + h
    attention(hbuf, LYR(l, 0), LYR(l, 1), w16[l][0], w16[l][1], w16[l][2], Bbuf);
    // y = LN(y, pn1) in place
    ln(Bbuf, LYR(l, 5), LYR(l, 6), Bbuf, nullptr);
    // z = attn2(LN(y, ln2)) + y
    attention(Bbuf, LYR(l, 7), LYR(l, 8), w16[l][3], w16[l][4], w16[l][5], Cbuf);
    // z = LN(z, pn2) in place (+ f16 for FF input)
    ln(Cbuf, LYR(l, 12), LYR(l, 13), Cbuf, ln16);
    // ffh = relu(z @ W1 + b1)  (f16)
    gemm(ln16, w16[l][6], LYR(l, 15), nullptr, nullptr, ffh16, nullptr, kFF, kC, 1);
    // a = ffh @ W2 + b2 + z -> hbuf
    gemm(ffh16, w16[l][7], LYR(l, 17), Cbuf, hbuf, nullptr, nullptr, kC, kFF, 0);
    // h = LN(a, pnl) in place
    ln(hbuf, LYR(l, 18), LYR(l, 19), hbuf, nullptr);
  }

  // logits = LN(h, lnf) @ Whead + bhead
  ln(hbuf, lnf_g, lnf_b, nullptr, ln16);
  gemm(ln16, whead16, bhead, nullptr, (float*)d_out, nullptr, nullptr, kV, kC, 0);
}
